// VQVAE_28269474742911
// MI455X (gfx1250) — matrix operand order repaired and checked by reading, not compiled, not run on hardware
//
#include <hip/hip_runtime.h>
#include <hip/hip_bf16.h>

// Problem constants (fixed by the reference): B=8, N=2048, D=256, K=8192
#define M_TOT   16384          // B*N tokens
#define D_DIM   256
#define K_CODES 8192

#define WAVES_PER_BLOCK 8
#define KCHUNK          32     // codes staged in LDS per pipeline stage
#define NCHUNK          (K_CODES / KCHUNK)
#define CB_STRIDE       264    // 256 + 8 bf16 pad -> conflict-free ds_load_b128

#ifndef __has_builtin
#define __has_builtin(x) 0
#endif
#if __has_builtin(__builtin_amdgcn_global_load_async_to_lds_b128) && \
    __has_builtin(__builtin_amdgcn_s_wait_asynccnt)
#define USE_ASYNC 1
#else
#define USE_ASYNC 0
#endif

typedef __attribute__((ext_vector_type(16))) __bf16 v16bf;
typedef __attribute__((ext_vector_type(8)))  float  v8f;
typedef __attribute__((ext_vector_type(4)))  int    v4i;

typedef __attribute__((address_space(1))) v4i* gas_v4i_ptr;
typedef __attribute__((address_space(3))) v4i* las_v4i_ptr;

union Frag {
    v16bf v;
    uint4 q[2];
};

__device__ __forceinline__ unsigned short f32_to_bf16_rne(float f) {
    unsigned int x = __float_as_uint(f);
    unsigned int r = x + 0x7FFFu + ((x >> 16) & 1u);
    return (unsigned short)(r >> 16);
}
__device__ __forceinline__ float bf16_bits_to_f32(unsigned short h) {
    return __uint_as_float(((unsigned int)h) << 16);
}

// 16-byte global -> LDS copy: async (direct-to-LDS, ASYNCcnt) when available.
__device__ __forceinline__ void cp16_g2l(const unsigned short* g, unsigned short* l) {
#if USE_ASYNC
    __builtin_amdgcn_global_load_async_to_lds_b128(
        (gas_v4i_ptr)(g), (las_v4i_ptr)(l), 0, 0);
#else
    *(uint4*)l = *(const uint4*)g;
#endif
}

// ---------------------------------------------------------------------------
// Prep: split x into bf16 hi/lo, and emit z_e (= x) output copy.
__global__ __launch_bounds__(256) void split_x_kernel(const float* __restrict__ x,
                                                      unsigned short* __restrict__ xhi,
                                                      unsigned short* __restrict__ xlo,
                                                      float* __restrict__ out_ze,
                                                      int n) {
    for (int i = blockIdx.x * blockDim.x + threadIdx.x; i < n;
         i += gridDim.x * blockDim.x) {
        float f = x[i];
        unsigned short hi = f32_to_bf16_rne(f);
        float lo = f - bf16_bits_to_f32(hi);
        xhi[i] = hi;
        xlo[i] = f32_to_bf16_rne(lo);
        out_ze[i] = f;
    }
}

// Prep: split codebook into bf16 hi/lo.
__global__ __launch_bounds__(256) void split_cb_kernel(const float* __restrict__ cb,
                                                       unsigned short* __restrict__ chi,
                                                       unsigned short* __restrict__ clo,
                                                       int n) {
    for (int i = blockIdx.x * blockDim.x + threadIdx.x; i < n;
         i += gridDim.x * blockDim.x) {
        float f = cb[i];
        unsigned short hi = f32_to_bf16_rne(f);
        float lo = f - bf16_bits_to_f32(hi);
        chi[i] = hi;
        clo[i] = f32_to_bf16_rne(lo);
    }
}

// Prep: c2[k] = sum_d cb[k][d]^2 (exact f32). One wave32 per code.
__global__ __launch_bounds__(256) void c2_kernel(const float* __restrict__ cb,
                                                 float* __restrict__ c2) {
    int tid  = threadIdx.x;
    int lane = tid & 31;
    int code = blockIdx.x * (blockDim.x / 32) + (tid >> 5);
    const float4* p = (const float4*)(cb + code * D_DIM + lane * 8);
    float4 a = p[0];
    float4 b = p[1];
    float s = a.x * a.x + a.y * a.y + a.z * a.z + a.w * a.w +
              b.x * b.x + b.y * b.y + b.z * b.z + b.w * b.w;
    #pragma unroll
    for (int off = 16; off >= 1; off >>= 1)
        s += __shfl_xor(s, off, 32);
    if (lane == 0) c2[code] = s;
}

// ---------------------------------------------------------------------------
// Main fused kernel: bf16-split WMMA GEMM + argmin over K + gather, with a
// double-buffered, (async when available) global->LDS codebook pipeline.
__global__ __launch_bounds__(256) void vq_wmma_kernel(
        const unsigned short* __restrict__ xhi,
        const unsigned short* __restrict__ xlo,
        const unsigned short* __restrict__ chi,
        const unsigned short* __restrict__ clo,
        const float* __restrict__ c2,
        const float* __restrict__ codebook,
        float* __restrict__ out_recon,
        float* __restrict__ out_zq,
        float* __restrict__ out_idx) {
    __shared__ unsigned short lds_hi[2][KCHUNK * CB_STRIDE];
    __shared__ unsigned short lds_lo[2][KCHUNK * CB_STRIDE];

    const int tid  = threadIdx.x;
    const int wave = tid >> 5;
    const int lane = tid & 31;
    const int half = lane >> 4;   // 0: rows 0-7 of C-tile, 1: rows 8-15
    const int n16  = lane & 15;   // column within 16-wide tile
    const int rowBase = (blockIdx.x * WAVES_PER_BLOCK + wave) * 16;
    const int m = rowBase + n16;  // A-matrix row owned by this lane

    // A fragments (16x32 bf16 per k-step), hi and lo splits, resident all sweep.
    // Lane L (half=L/16, M=L%16): elems 0-7 -> K = 8*half+e ; elems 8-15 -> K = 16+8*half+(e-8)
    Frag a_hi[8], a_lo[8];
    #pragma unroll
    for (int s = 0; s < 8; ++s) {
        int d0 = s * 32 + 8 * half;
        int d1 = s * 32 + 16 + 8 * half;
        a_hi[s].q[0] = *(const uint4*)(xhi + m * D_DIM + d0);
        a_hi[s].q[1] = *(const uint4*)(xhi + m * D_DIM + d1);
        a_lo[s].q[0] = *(const uint4*)(xlo + m * D_DIM + d0);
        a_lo[s].q[1] = *(const uint4*)(xlo + m * D_DIM + d1);
    }

    float best[8];
    int   bidx[8];
    #pragma unroll
    for (int v = 0; v < 8; ++v) { best[v] = 3.4e38f; bidx[v] = 0; }

    // Stage chunk 0 into buffer 0 (8 x 16B per thread, hi+lo).
    #pragma unroll
    for (int i = tid; i < KCHUNK * (D_DIM / 8); i += 256) {
        int code  = i >> 5;          // 32 uint4 per code row
        int dpart = (i & 31) << 3;   // 8 bf16 per uint4
        cp16_g2l(chi + code * D_DIM + dpart, &lds_hi[0][code * CB_STRIDE + dpart]);
        cp16_g2l(clo + code * D_DIM + dpart, &lds_lo[0][code * CB_STRIDE + dpart]);
    }

    for (int c = 0; c < NCHUNK; ++c) {
        const int cur = c & 1;
        const int kc  = c * KCHUNK;
        if (c + 1 < NCHUNK) {
            // Prefetch chunk c+1 into the other buffer (its readers finished
            // before the trailing barrier of iteration c-1).
            const int kn = kc + KCHUNK;
            #pragma unroll
            for (int i = tid; i < KCHUNK * (D_DIM / 8); i += 256) {
                int code  = i >> 5;
                int dpart = (i & 31) << 3;
                cp16_g2l(chi + (kn + code) * D_DIM + dpart,
                         &lds_hi[cur ^ 1][code * CB_STRIDE + dpart]);
                cp16_g2l(clo + (kn + code) * D_DIM + dpart,
                         &lds_lo[cur ^ 1][code * CB_STRIDE + dpart]);
            }
#if USE_ASYNC
            // Async loads complete in order: <=8 outstanding => chunk c done.
            __builtin_amdgcn_s_wait_asynccnt(8);
#endif
        } else {
#if USE_ASYNC
            __builtin_amdgcn_s_wait_asynccnt(0);
#endif
        }
        __syncthreads();  // all waves' chunk-c staging visible

        #pragma unroll
        for (int ct = 0; ct < KCHUNK; ct += 16) {
            v8f acc = {};
            // B fragment lane view: col = n16, elems 0-7 -> K = 8*half+e ; elems 8-15 -> K = 16+8*half+(e-8)
            const unsigned short* bh = &lds_hi[cur][(ct + n16) * CB_STRIDE + 8 * half];
            const unsigned short* bl = &lds_lo[cur][(ct + n16) * CB_STRIDE + 8 * half];
            #pragma unroll
            for (int s = 0; s < 8; ++s) {
                Frag bhf, blf;
                bhf.q[0] = *(const uint4*)(bh + s * 32);
                bhf.q[1] = *(const uint4*)(bh + s * 32 + 16);
                blf.q[0] = *(const uint4*)(bl + s * 32);
                blf.q[1] = *(const uint4*)(bl + s * 32 + 16);
                acc = __builtin_amdgcn_wmma_f32_16x16x32_bf16(
                    false, a_hi[s].v, false, bhf.v, (short)0, acc, false, false);
                acc = __builtin_amdgcn_wmma_f32_16x16x32_bf16(
                    false, a_hi[s].v, false, blf.v, (short)0, acc, false, false);
                acc = __builtin_amdgcn_wmma_f32_16x16x32_bf16(
                    false, a_lo[s].v, false, bhf.v, (short)0, acc, false, false);
            }
            // score = c2[col] - 2*cross  (z2 is row-constant, drops out of argmin)
            int colg = kc + ct + n16;
            float cc = c2[colg];
            #pragma unroll
            for (int v = 0; v < 8; ++v) {
                float cand = fmaf(-2.0f, acc[v], cc);
                if (cand < best[v]) { best[v] = cand; bidx[v] = colg; }
            }
        }
        __syncthreads();  // all reads of buffer `cur` done before it is restaged
    }

    // Per-row reduction across the 16 lanes of each half; tie -> lower index
    // (matches jnp.argmin first-occurrence semantics).
    #pragma unroll
    for (int v = 0; v < 8; ++v) {
        float bv = best[v];
        int   bi = bidx[v];
        #pragma unroll
        for (int off = 8; off >= 1; off >>= 1) {
            float ov = __shfl_xor(bv, off, 32);
            int   oi = __shfl_xor(bi, off, 32);
            if (ov < bv || (ov == bv && oi < bi)) { bv = ov; bi = oi; }
        }
        int row = rowBase + 8 * half + v;
        if (n16 == 0) out_idx[row] = (float)bi;
        // 16 lanes of this half cooperatively gather the winning f32 code row.
        const float4* src = (const float4*)(codebook + bi * D_DIM + n16 * 16);
        float4* dq = (float4*)(out_zq + row * D_DIM + n16 * 16);
        float4* dr = (float4*)(out_recon + row * D_DIM + n16 * 16);
        #pragma unroll
        for (int j = 0; j < 4; ++j) {
            float4 t = src[j];
            dq[j] = t;
            dr[j] = t;
        }
    }
}

// ---------------------------------------------------------------------------
extern "C" void kernel_launch(void* const* d_in, const int* in_sizes, int n_in,
                              void* d_out, int out_size, void* d_ws, size_t ws_size,
                              hipStream_t stream) {
    const float* x  = (const float*)d_in[0];   // (8,2048,256) f32
    const float* cb = (const float*)d_in[1];   // (8192,256)  f32

    const int n_x  = M_TOT * D_DIM;            // 4,194,304
    const int n_cb = K_CODES * D_DIM;          // 2,097,152

    // Workspace layout (bytes): xhi 8MB | xlo 8MB | chi 4MB | clo 4MB | c2 32KB
    char* ws = (char*)d_ws;
    unsigned short* xhi = (unsigned short*)ws;
    unsigned short* xlo = xhi + n_x;
    unsigned short* chi = xlo + n_x;
    unsigned short* clo = chi + n_cb;
    float*          c2  = (float*)(clo + n_cb);

    // Output layout (floats): x_recon | z_e | z_q | indices(as float)
    float* out       = (float*)d_out;
    float* out_recon = out;
    float* out_ze    = out + n_x;
    float* out_zq    = out + 2 * n_x;
    float* out_idx   = out + 3 * n_x;

    split_x_kernel<<<2048, 256, 0, stream>>>(x, xhi, xlo, out_ze, n_x);
    split_cb_kernel<<<1024, 256, 0, stream>>>(cb, chi, clo, n_cb);
    c2_kernel<<<K_CODES / (256 / 32), 256, 0, stream>>>(cb, c2);

    const int blocks = M_TOT / (16 * WAVES_PER_BLOCK);  // 128
    vq_wmma_kernel<<<blocks, 256, 0, stream>>>(xhi, xlo, chi, clo, c2, cb,
                                               out_recon, out_zq, out_idx);
}
